// MultiHeadAttention_71210557768100
// MI455X (gfx1250) — compile-verified, matrix operand order checked by reading, not run on hardware
//
#include <hip/hip_runtime.h>
#include <hip/hip_bf16.h>
#include <stdint.h>

// ---------------------------------------------------------------------------
// MultiHeadAttention (channel-token attention) + residual + BatchNorm3d
// b=16, c=256, hwd=4096, n_head=8, ldim=256, hp=2048
// All large GEMMs via v_wmma_f32_16x16x32_bf16 (fp32 accumulate).
// WG tile 128x128x32, 8 waves, wave tile 64x32 (8 WMMA / 12 ds_load_b128),
// double-buffered LDS with register prefetch (1 barrier per K-step).
// ---------------------------------------------------------------------------

typedef __attribute__((ext_vector_type(16))) __bf16 v16bf;
typedef __attribute__((ext_vector_type(8)))  float  v8f;

#define BATCH   16
#define CHN     256
#define HWD     4096
#define NHEAD   8
#define LDIM    256
#define HP      2048   // NHEAD*LDIM
#define INV_TEMP (1.0f/64.0f)   // 1/sqrt(4096)

__device__ __forceinline__ unsigned short f2bf(float f) {
  unsigned int u = __float_as_uint(f);
  unsigned int r = (u + 0x7FFFu + ((u >> 16) & 1u)) >> 16;  // round-to-nearest-even
  return (unsigned short)r;
}

enum { OM_F32 = 0, OM_F32_RES = 1, OM_BF16 = 2, OM_BF16_T = 3 };

union Frag { uint4 u4[2]; v16bf v; };

constexpr int BM = 128, BN = 128, BK = 32, LDP = 40;  // LDP: padded bf16 stride

// Load one 128x32 tile's worth (16 elems/thread) from global into regs as bf16.
template<bool BF16>
__device__ __forceinline__ void ld_tile(const void* __restrict__ p, int r0, int ld,
                                        int kt, int t, uint2 (&d)[4]) {
#pragma unroll
  for (int it = 0; it < 4; ++it) {
    int idx = (it * 256 + t) * 4;
    int r = idx >> 5, cc = idx & 31;
    if (BF16) {
      d[it] = *(const uint2*)((const unsigned short*)p + (size_t)(r0 + r) * ld + kt + cc);
    } else {
      float4 f = *(const float4*)((const float*)p + (size_t)(r0 + r) * ld + kt + cc);
      d[it].x = (unsigned)f2bf(f.x) | ((unsigned)f2bf(f.y) << 16);
      d[it].y = (unsigned)f2bf(f.z) | ((unsigned)f2bf(f.w) << 16);
    }
  }
}

__device__ __forceinline__ void st_tile(unsigned short* __restrict__ s, int t,
                                        const uint2 (&d)[4]) {
#pragma unroll
  for (int it = 0; it < 4; ++it) {
    int idx = (it * 256 + t) * 4;
    int r = idx >> 5, cc = idx & 31;
    *(uint2*)&s[r * LDP + cc] = d[it];
  }
}

// C[M,N] = alpha * A[M,K] x W[N,K]^T  (+ residual), batched over blockIdx.z.
template<bool A_BF16, bool W_BF16, int OMODE>
__global__ __launch_bounds__(256)
void gemm_wmma(const void* __restrict__ Ap, const void* __restrict__ Wp,
               void* __restrict__ Cp, const float* __restrict__ Res,
               int Ksz, int lda, int ldw, int ldc,
               long oA_hi, long oA_lo, long oW_hi, long oW_lo,
               long oC_hi, long oC_lo, int zlo, float alpha)
{
  __shared__ unsigned short As[2][BM * LDP];   // 2 x 10240 B
  __shared__ unsigned short Bs[2][BN * LDP];   // 2 x 10240 B

  const int t    = threadIdx.x;
  const int lane = t & 31;
  const int wid  = t >> 5;
  const int wm   = wid & 1;           // 0..1  (M direction, 2 x 64)
  const int wn   = wid >> 1;          // 0..3  (N direction, 4 x 32)
  const int row0 = blockIdx.y * BM;
  const int col0 = blockIdx.x * BN;

  const long z   = (long)blockIdx.z;
  const long zh  = z / zlo, zl = z % zlo;
  const long offA = zh * oA_hi + zl * oA_lo;
  const long offW = zh * oW_hi + zl * oW_lo;
  const long offC = zh * oC_hi + zl * oC_lo;

  const void* Az = A_BF16 ? (const void*)((const unsigned short*)Ap + offA)
                          : (const void*)((const float*)Ap + offA);
  const void* Wz = W_BF16 ? (const void*)((const unsigned short*)Wp + offW)
                          : (const void*)((const float*)Wp + offW);
  float*          Cf = (float*)Cp + offC;
  unsigned short* Cb = (unsigned short*)Cp + offC;
  const float*    Rp = (OMODE == OM_F32_RES) ? (Res + offC) : nullptr;

  v8f acc[4][2];
#pragma unroll
  for (int i = 0; i < 4; ++i)
#pragma unroll
    for (int j = 0; j < 2; ++j) acc[i][j] = (v8f)0.0f;

  const int lrow = lane & 15;   // row within 16x16 fragment
  const int lsel = lane >> 4;   // K-half select

  const int nk = Ksz / BK;
  uint2 pa[4], pb[4];

  // prologue: stage tile 0
  ld_tile<A_BF16>(Az, row0, lda, 0, t, pa);
  ld_tile<W_BF16>(Wz, col0, ldw, 0, t, pb);
  st_tile(&As[0][0], t, pa);
  st_tile(&Bs[0][0], t, pb);
  __syncthreads();

  int cur = 0;
  for (int ki = 0; ki < nk; ++ki) {
    const bool more = (ki + 1 < nk);
    if (more) {  // prefetch next tile into regs (overlaps with WMMAs below)
      ld_tile<A_BF16>(Az, row0, lda, (ki + 1) * BK, t, pa);
      ld_tile<W_BF16>(Wz, col0, ldw, (ki + 1) * BK, t, pb);
    }

    // fragments (ISA 16-bit A/B layout: lane%16 = row, lane/16 = K half)
    const unsigned short* Ac = &As[cur][0];
    const unsigned short* Bc = &Bs[cur][0];
    Frag b[2];
#pragma unroll
    for (int j = 0; j < 2; ++j) {
      int br = wn * 32 + j * 16 + lrow;
      b[j].u4[0] = *(const uint4*)&Bc[br * LDP + lsel * 8];
      b[j].u4[1] = *(const uint4*)&Bc[br * LDP + 16 + lsel * 8];
    }
#pragma unroll
    for (int i = 0; i < 4; ++i) {
      int ar = wm * 64 + i * 16 + lrow;
      Frag a;
      a.u4[0] = *(const uint4*)&Ac[ar * LDP + lsel * 8];
      a.u4[1] = *(const uint4*)&Ac[ar * LDP + 16 + lsel * 8];
#pragma unroll
      for (int j = 0; j < 2; ++j)
        acc[i][j] = __builtin_amdgcn_wmma_f32_16x16x32_bf16(
            false, a.v, false, b[j].v, (short)0, acc[i][j], false, false);
    }

    if (more) {  // fill the other buffer
      st_tile(&As[cur ^ 1][0], t, pa);
      st_tile(&Bs[cur ^ 1][0], t, pb);
    }
    __syncthreads();
    cur ^= 1;
  }

  // epilogue: C layout lane%16 = N, vgpr r + 8*(lane/16) = M
#pragma unroll
  for (int i = 0; i < 4; ++i) {
#pragma unroll
    for (int j = 0; j < 2; ++j) {
      int m0 = row0 + wm * 64 + i * 16 + (lane >> 4) * 8;
      int n  = col0 + wn * 32 + j * 16 + (lane & 15);
#pragma unroll
      for (int r = 0; r < 8; ++r) {
        int m = m0 + r;
        float val = acc[i][j][r] * alpha;
        if (OMODE == OM_F32) {
          Cf[(size_t)m * ldc + n] = val;
        } else if (OMODE == OM_F32_RES) {
          Cf[(size_t)m * ldc + n] = val + Rp[(size_t)m * ldc + n];
        } else if (OMODE == OM_BF16) {
          Cb[(size_t)m * ldc + n] = f2bf(val);
        } else { // OM_BF16_T : [b*256+cc, head*256+l] -> Vt[(b*8+head)][l][cc]
          size_t idx = ((size_t)((m >> 8) * 8 + (n >> 8)) << 16) +
                       ((size_t)(n & 255) << 8) + (size_t)(m & 255);
          Cb[idx] = f2bf(val);
        }
      }
    }
  }
}

// one wave per 256-wide row: softmax -> bf16 probs
__global__ __launch_bounds__(256)
void softmax_rows(const float* __restrict__ S, unsigned short* __restrict__ P)
{
  const int lane = threadIdx.x & 31;
  const int wid  = threadIdx.x >> 5;
  const size_t row = (size_t)blockIdx.x * 8 + wid;
  const float* p = S + row * 256;

  float x[8];
  float mx = -3.4e38f;
#pragma unroll
  for (int i = 0; i < 8; ++i) { x[i] = p[lane + i * 32]; mx = fmaxf(mx, x[i]); }
#pragma unroll
  for (int o = 16; o > 0; o >>= 1) mx = fmaxf(mx, __shfl_xor(mx, o, 32));
  float s = 0.0f;
#pragma unroll
  for (int i = 0; i < 8; ++i) { x[i] = __expf(x[i] - mx); s += x[i]; }
#pragma unroll
  for (int o = 16; o > 0; o >>= 1) s += __shfl_xor(s, o, 32);
  const float inv = 1.0f / s;
  unsigned short* q = P + row * 256;
#pragma unroll
  for (int i = 0; i < 8; ++i) q[lane + i * 32] = f2bf(x[i] * inv);
}

// one WG per channel: mean/var over (b, h*w*d) = 65536 elems
__global__ __launch_bounds__(256)
void bn_stats(const float* __restrict__ X, float* __restrict__ mv)
{
  const int ch = blockIdx.x;
  const int t  = threadIdx.x;
  float s = 0.0f, ss = 0.0f;
  for (int b = 0; b < BATCH; ++b) {
    const float* p = X + ((size_t)b * CHN + ch) * HWD;
    for (int i = t; i < HWD; i += 256) { float v = p[i]; s += v; ss += v * v; }
  }
  __shared__ float sh0[256], sh1[256];
  sh0[t] = s; sh1[t] = ss;
  __syncthreads();
  for (int o = 128; o > 0; o >>= 1) {
    if (t < o) { sh0[t] += sh0[t + o]; sh1[t] += sh1[t + o]; }
    __syncthreads();
  }
  if (t == 0) {
    float m = sh0[0] * (1.0f / 65536.0f);
    float v = sh1[0] * (1.0f / 65536.0f) - m * m;
    mv[ch] = m; mv[CHN + ch] = v;
  }
}

__global__ __launch_bounds__(256)
void bn_apply(float* __restrict__ X, const float* __restrict__ mv,
              const float* __restrict__ gamma, const float* __restrict__ beta)
{
  const size_t i = (size_t)blockIdx.x * 256 + threadIdx.x;
  const int ch = (int)((i >> 12) & 255);
  const float m = mv[ch];
  const float v = mv[CHN + ch];
  X[i] = (X[i] - m) * rsqrtf(v + 1e-5f) * gamma[ch] + beta[ch];
}

extern "C" void kernel_launch(void* const* d_in, const int* in_sizes, int n_in,
                              void* d_out, int out_size, void* d_ws, size_t ws_size,
                              hipStream_t stream)
{
  (void)in_sizes; (void)n_in; (void)out_size; (void)ws_size;
  const float* v_in  = (const float*)d_in[0];
  const float* k_in  = (const float*)d_in[1];
  const float* q_in  = (const float*)d_in[2];
  const float* Wq    = (const float*)d_in[3];
  const float* Wk    = (const float*)d_in[4];
  const float* Wv    = (const float*)d_in[5];
  const float* Wfc   = (const float*)d_in[6];
  const float* gamma = (const float*)d_in[7];
  const float* beta  = (const float*)d_in[8];
  float* out = (float*)d_out;

  // workspace layout (bytes)
  char* ws = (char*)d_ws;
  unsigned short* Qb = (unsigned short*)(ws + 0);                  // [4096,2048] bf16
  unsigned short* Kb = (unsigned short*)(ws + (size_t)16777216);
  unsigned short* Vt = (unsigned short*)(ws + (size_t)33554432);   // [128][256][256]
  float*          Sc = (float*)        (ws + (size_t)50331648);    // [128][256][256] f32
  unsigned short* Pr = (unsigned short*)(ws + (size_t)83886080);   // probs bf16
  unsigned short* AO = (unsigned short*)(ws + (size_t)100663296);  // [4096,2048] bf16
  float*          MV = (float*)        (ws + (size_t)117440512);   // mean/var [2][256]

  const int MBC = BATCH * CHN;  // 4096 rows (b*c)

  // 1) Q = (qf @ Wq^T) * (1/TEMP)   -> bf16 [4096, 2048]
  gemm_wmma<false, false, OM_BF16><<<dim3(HP / BN, MBC / BM, 1), 256, 0, stream>>>(
      q_in, Wq, Qb, nullptr, HWD, HWD, HWD, HP, 0, 0, 0, 0, 0, 0, 1, INV_TEMP);
  // 2) K = kf @ Wk^T                -> bf16 [4096, 2048]
  gemm_wmma<false, false, OM_BF16><<<dim3(HP / BN, MBC / BM, 1), 256, 0, stream>>>(
      k_in, Wk, Kb, nullptr, HWD, HWD, HWD, HP, 0, 0, 0, 0, 0, 0, 1, 1.0f);
  // 3) V = vf @ Wv^T, stored transposed per (b,head): Vt[(b*8+h)][ldim][c]
  gemm_wmma<false, false, OM_BF16_T><<<dim3(HP / BN, MBC / BM, 1), 256, 0, stream>>>(
      v_in, Wv, Vt, nullptr, HWD, HWD, HWD, 0, 0, 0, 0, 0, 0, 0, 1, 1.0f);

  // 4) scores[z][c][c] = Q_z @ K_z^T, z = b*8+head (Q already scaled)
  gemm_wmma<true, true, OM_F32><<<dim3(CHN / BN, CHN / BM, BATCH * NHEAD), 256, 0, stream>>>(
      Qb, Kb, Sc, nullptr, LDIM, HP, HP, CHN,
      (long)CHN * HP, LDIM,                       // A: per-b, per-head offsets
      (long)CHN * HP, LDIM,                       // W: same
      (long)NHEAD * CHN * CHN, (long)CHN * CHN,   // C: z-contiguous
      NHEAD, 1.0f);

  // 5) softmax rows -> bf16 probs
  softmax_rows<<<dim3((BATCH * NHEAD * CHN) / 8), 256, 0, stream>>>(Sc, Pr);

  // 6) AO[b*256+i][head*256+l] = probs_z @ Vt_z^T
  gemm_wmma<true, true, OM_BF16><<<dim3(LDIM / BN, CHN / BM, BATCH * NHEAD), 256, 0, stream>>>(
      Pr, Vt, AO, nullptr, CHN, CHN, CHN, HP,
      (long)NHEAD * CHN * CHN, (long)CHN * CHN,   // A: probs z-contiguous
      (long)NHEAD * CHN * CHN, (long)CHN * CHN,   // W: Vt z-contiguous
      (long)CHN * HP, LDIM,                       // C: per-b, per-head offsets
      NHEAD, 1.0f);

  // 7) out = vf + AO @ Wfc^T  -> fp32 d_out [4096, 4096]
  gemm_wmma<true, false, OM_F32_RES><<<dim3(HWD / BN, MBC / BM, 1), 256, 0, stream>>>(
      AO, Wfc, out, v_in, HP, HP, HP, HWD, 0, 0, 0, 0, 0, 0, 1, 1.0f);

  // 8) BatchNorm3d (batch statistics), in-place on d_out
  bn_stats<<<dim3(CHN), 256, 0, stream>>>(out, MV);
  bn_apply<<<dim3((size_t)BATCH * CHN * HWD / 256), 256, 0, stream>>>(out, MV, gamma, beta);
}
